// HPINet_11888469475645
// MI455X (gfx1250) — matrix operand order repaired and checked by reading, not compiled, not run on hardware
//
#include <hip/hip_runtime.h>
#include <hip/hip_bf16.h>
#include <math.h>

// ---------------------------------------------------------------------------
// Types for CDNA5 WMMA (wave32)
// ---------------------------------------------------------------------------
typedef _Float16 half_t;
typedef __attribute__((ext_vector_type(16))) _Float16 v16h;
typedef __attribute__((ext_vector_type(8)))  float    v8f;

#define DEV __device__ __forceinline__

DEV int lane_id() { return (int)(threadIdx.x & 31u); }

DEV v8f wmma16(v16h a, v16h b, v8f c) {
  // D = A(16x32 f16) * B(32x16 f16) + C(16x16 f32)
  return __builtin_amdgcn_wmma_f32_16x16x32_f16(false, a, false, b, (short)0, c,
                                                false, false);
}

union FragU { uint4 u[2]; v16h h; };

// A fragment: 16 rows x 32 K f16 from row-major [rows, lda] at (t0, k0).
// lane m=l&15 ; high half-wave K base +16 (k = 16h + e).
// Per lane the fragment is one contiguous 16-half (32B) run -> two b128 loads.
DEV v16h afrag_load(const half_t* A, long t0, int lda, int k0) {
  int l = lane_id();
  const half_t* row = A + (t0 + (l & 15)) * (long)lda + k0 + ((l >> 4) << 4);
  FragU f;
  f.u[0] = *(const uint4*)(row);        // e 0..7  == k +0..7
  f.u[1] = *(const uint4*)(row + 8);    // e 8..15 == k +8..15
  return f.h;
}

// B fragment from pre-swizzled panel [kt][ot][lane][16] -> 32B contiguous/lane.
DEV v16h bfrag_swz(const half_t* W, int kt, int ot, int OT) {
  int l = lane_id();
  const half_t* p = W + ((size_t)(kt * OT + ot) * 32 + l) * 16;
  FragU f;
  f.u[0] = *(const uint4*)(p);
  f.u[1] = *(const uint4*)(p + 8);
  return f.h;
}

// B fragment for K^T (scores): B[k][n] = Kb[(key0+n)*32 + k].
// Per lane: 16 contiguous halves -> two b128 loads.
DEV v16h bfrag_kt(const half_t* Kb, long key0) {
  int l = lane_id();
  const half_t* row = Kb + (key0 + (l & 15)) * 32 + ((l >> 4) << 4);
  FragU f;
  f.u[0] = *(const uint4*)(row);
  f.u[1] = *(const uint4*)(row + 8);
  return f.h;
}

// ---------------------------------------------------------------------------
// Generic token GEMM: one wave computes one 16x16 output tile.
// A: [nT, lda] f16 ; W: swizzled panel ; optional bias.
// act==1 -> exact GELU.  Writes f16 [nT, ldh] and/or accumulates f32 [nT, ldadd].
// ---------------------------------------------------------------------------
__global__ __launch_bounds__(32) void gemm_tok_kernel(
    const half_t* __restrict__ A, int lda,
    const half_t* __restrict__ W,
    const float* __restrict__ bias,
    half_t* __restrict__ outH, int ldh,
    float* __restrict__ outAdd, int ldadd,
    long nT, int ktiles, int O, int otiles, int act)
{
  long gid = blockIdx.x;
  int  ot  = (int)(gid % otiles);
  long t0  = (gid / otiles) * 16;
  v8f acc = {};
  for (int kt = 0; kt < ktiles; ++kt) {
    v16h a = afrag_load(A, t0, lda, kt * 32);
    v16h b = bfrag_swz(W, kt, ot, otiles);
    acc = wmma16(a, b, acc);
  }
  int l = lane_id();
  int col = ot * 16 + (l & 15);
  float bv = (bias != nullptr && col < O) ? bias[col] : 0.0f;
  int rbase = (l >> 4) << 3;
#pragma unroll
  for (int r = 0; r < 8; ++r) {
    long t = t0 + rbase + r;
    float v = acc[r] + bv;
    if (act == 1) v = 0.5f * v * (1.0f + erff(v * 0.70710678f));
    if (outH) outH[t * ldh + col] = (half_t)v;
    if (outAdd && col < O) outAdd[t * ldadd + col] += v;
  }
}

// ---------------------------------------------------------------------------
// Repack V [n,T,48] f16 into B-fragment-native chunks:
// vs[((p*chunks + ch)*3 + ct)*512 + lane*16 + e] = V[key][ct*16 + (lane&15)]
//   with key = ch*32 + (lane>=16)*16 + e ; zero-padded past T.
// ---------------------------------------------------------------------------
__global__ void repack_v_kernel(const half_t* __restrict__ vh, half_t* __restrict__ vs,
                                int T, int chunks, long total)
{
  long i = (long)blockIdx.x * blockDim.x + threadIdx.x;
  if (i >= total) return;
  int e    = (int)(i & 15);
  int lane = (int)((i >> 4) & 31);
  long rest = i >> 9;
  int ct = (int)(rest % 3);
  long rest2 = rest / 3;
  int ch = (int)(rest2 % chunks);
  int p  = (int)(rest2 / chunks);
  int key = ch * 32 + ((lane >> 4) << 4) + e;
  half_t v = (half_t)0.0f;
  if (key < T) v = vh[((long)p * T + key) * 48 + ct * 16 + (lane & 15)];
  vs[i] = v;
}

// ---------------------------------------------------------------------------
// Flash attention per (patch, 16-query tile). T is a multiple of 16.
// Q,K: [n,T,32] f16 ; Vs: swizzled [n, chunks,3,32,16] f16 ; X: [n,T,40] f32.
// ---------------------------------------------------------------------------
__global__ __launch_bounds__(32) void attn_kernel(
    const half_t* __restrict__ Qh, const half_t* __restrict__ Kh,
    const half_t* __restrict__ Vs, float* __restrict__ X, int T, int chunks)
{
  int qtiles = T >> 4;
  int patch  = blockIdx.x / qtiles;
  int qloc   = (blockIdx.x % qtiles) * 16;
  const half_t* Q = Qh + (long)patch * T * 32;
  const half_t* K = Kh + (long)patch * T * 32;
  const half_t* V = Vs + (long)patch * chunks * 1536;
  float* Xp = X + (long)patch * T * 40;

  __shared__ half_t Plds[16 * 32];
  int l = lane_id();
  v16h qa = afrag_load(Q, qloc, 32, 0);
  v8f o0 = {}, o1 = {}, o2 = {};
  float mrow[8], lrow[8];
#pragma unroll
  for (int r = 0; r < 8; ++r) { mrow[r] = -3.0e38f; lrow[r] = 0.0f; }
  const float scale = 0.20412415f;  // 24^-0.5
  int prow = (l >> 4) << 3;

  for (int k0 = 0; k0 < T; k0 += 32) {
    int c = k0 >> 5;
    if (k0 + 32 < T) {  // gfx1250 global_prefetch of the next K/V chunk
      __builtin_prefetch((const void*)(K + (long)(k0 + 32) * 32), 0, 1);
      __builtin_prefetch((const void*)(V + (long)(c + 1) * 1536), 0, 1);
    }
    v8f s0 = {}, s1 = {};
    s0 = wmma16(qa, bfrag_kt(K, k0), s0);
    s1 = wmma16(qa, bfrag_kt(K, k0 + 16), s1);
    bool v1 = (k0 + 16) < T;
#pragma unroll
    for (int r = 0; r < 8; ++r) {
      float a0 = s0[r] * scale;
      float a1 = v1 ? s1[r] * scale : -1.0e30f;
      float mx = fmaxf(a0, a1);
#pragma unroll
      for (int off = 1; off < 16; off <<= 1) mx = fmaxf(mx, __shfl_xor(mx, off, 16));
      float mnew  = fmaxf(mrow[r], mx);
      float alpha = __expf(mrow[r] - mnew);
      mrow[r] = mnew;
      float p0 = __expf(a0 - mnew);
      float p1 = __expf(a1 - mnew);
      float rs = p0 + p1;
#pragma unroll
      for (int off = 1; off < 16; off <<= 1) rs += __shfl_xor(rs, off, 16);
      lrow[r] = lrow[r] * alpha + rs;
      o0[r] *= alpha; o1[r] *= alpha; o2[r] *= alpha;
      Plds[(prow + r) * 32 + (l & 15)]      = (half_t)p0;
      Plds[(prow + r) * 32 + 16 + (l & 15)] = (half_t)p1;
    }
    __syncthreads();
    v16h pa = afrag_load(Plds, 0, 32, 0);
    o0 = wmma16(pa, bfrag_swz(V, c, 0, 3), o0);
    o1 = wmma16(pa, bfrag_swz(V, c, 1, 3), o1);
    o2 = wmma16(pa, bfrag_swz(V, c, 2, 3), o2);
    __syncthreads();
  }
  int col = l & 15;
#pragma unroll
  for (int r = 0; r < 8; ++r) {
    long t = qloc + prow + r;
    float inv = 1.0f / lrow[r];
    Xp[t * 40 + col]      += o0[r] * inv;
    Xp[t * 40 + 16 + col] += o1[r] * inv;
    if (col < 8) Xp[t * 40 + 32 + col] += o2[r] * inv;
  }
}

// ---------------------------------------------------------------------------
// LayerNorm over 40 channels -> zero-padded f16 [nT, 64]
// ---------------------------------------------------------------------------
__global__ void ln_kernel(const float* __restrict__ X, const float* __restrict__ g,
                          const float* __restrict__ b, half_t* __restrict__ out, long nT)
{
  long t = (long)blockIdx.x * blockDim.x + threadIdx.x;
  if (t >= nT) return;
  const float* row = X + t * 40;
  float m = 0.0f;
#pragma unroll
  for (int c = 0; c < 40; ++c) m += row[c];
  m *= 0.025f;
  float v = 0.0f;
#pragma unroll
  for (int c = 0; c < 40; ++c) { float d = row[c] - m; v += d * d; }
  v *= 0.025f;
  float rstd = rsqrtf(v + 1e-5f);
  half_t* o = out + t * 64;
#pragma unroll
  for (int c = 0; c < 40; ++c) o[c] = (half_t)((row[c] - m) * rstd * g[c] + b[c]);
#pragma unroll
  for (int c = 40; c < 64; ++c) o[c] = (half_t)0.0f;
}

// ---------------------------------------------------------------------------
// Patch divide / match / gather / reverse helpers
// ---------------------------------------------------------------------------
__global__ void patch_divide_kernel(const float* __restrict__ f, float* __restrict__ xtok,
                                    int ps, int step, int nw, long total)
{
  long i = (long)blockIdx.x * blockDim.x + threadIdx.x;
  if (i >= total) return;
  int c = (int)(i % 40);
  long t = i / 40;
  int T = ps * ps;
  int tin = (int)(t % T);
  int p = (int)(t / T);
  int py = tin / ps, px = tin % ps;
  int pi = p / nw, pj = p % nw;
  int top = pi * step;  if (top + ps > 320)  top = 320 - ps;
  int left = pj * step; if (left + ps > 320) left = 320 - ps;
  xtok[i] = f[((long)c * 320 + top + py) * 320 + left + px];
}

__global__ void patch_mean_kernel(const float* __restrict__ xtok, float* __restrict__ means,
                                  int T, int total)
{
  int i = blockIdx.x * blockDim.x + threadIdx.x;
  if (i >= total) return;
  int p = i / 40, c = i % 40;
  float s = 0.0f;
  for (int t = 0; t < T; ++t) s += xtok[((long)p * T + t) * 40 + c];
  means[i] = s / (float)T;
}

__global__ void match_ln_kernel(const float* __restrict__ means, const float* __restrict__ g,
                                const float* __restrict__ b, float* __restrict__ qln, int n)
{
  int p = blockIdx.x * blockDim.x + threadIdx.x;
  if (p >= n) return;
  const float* row = means + p * 40;
  float m = 0.0f;
  for (int c = 0; c < 40; ++c) m += row[c];
  m *= 0.025f;
  float v = 0.0f;
  for (int c = 0; c < 40; ++c) { float d = row[c] - m; v += d * d; }
  v *= 0.025f;
  float rstd = rsqrtf(v + 1e-5f);
  for (int c = 0; c < 40; ++c) qln[p * 40 + c] = (row[c] - m) * rstd * g[c] + b[c];
}

__global__ void match_argmax_kernel(const float* __restrict__ qln, int* __restrict__ idx, int n)
{
  int i = blockIdx.x;
  int tid = threadIdx.x;
  const float* qi = qln + i * 40;
  float best = -3.4e38f; int bm = 0x7fffffff;
  for (int m = tid; m < n; m += blockDim.x) {
    const float* qm = qln + m * 40;
    float s = 0.0f;
    for (int c = 0; c < 40; ++c) s += qi[c] * qm[c];
    s *= 0.15811388f;              // 40^-0.5
    if (m == i) s -= 100.0f;
    if (s > best || (s == best && m < bm)) { best = s; bm = m; }
  }
  __shared__ float bv[128];
  __shared__ int   bi[128];
  bv[tid] = best; bi[tid] = bm;
  __syncthreads();
  for (int s = 64; s > 0; s >>= 1) {
    if (tid < s) {
      if (bv[tid + s] > bv[tid] || (bv[tid + s] == bv[tid] && bi[tid + s] < bi[tid])) {
        bv[tid] = bv[tid + s]; bi[tid] = bi[tid + s];
      }
    }
    __syncthreads();
  }
  if (tid == 0) idx[i] = bi[0];
}

__global__ void gather_y_kernel(const float* __restrict__ xtok, const int* __restrict__ idx,
                                half_t* __restrict__ yh, int T, long total)
{
  long i = (long)blockIdx.x * blockDim.x + threadIdx.x;
  if (i >= total) return;
  int c = (int)(i % 64);
  long t = i / 64;
  int tin = (int)(t % T);
  int p = (int)(t / T);
  yh[i] = (c < 40) ? (half_t)xtok[((long)idx[p] * T + tin) * 40 + c] : (half_t)0.0f;
}

__global__ void patch_scatter_kernel(const float* __restrict__ xtok, float* __restrict__ resid,
                                     int ps, int step, int nw, long total)
{
  long i = (long)blockIdx.x * blockDim.x + threadIdx.x;
  if (i >= total) return;
  int c = (int)(i % 40);
  long t = i / 40;
  int T = ps * ps;
  int tin = (int)(t % T);
  int p = (int)(t / T);
  int py = tin / ps, px = tin % ps;
  int pi = p / nw, pj = p % nw;
  int top = pi * step;  if (top + ps > 320)  top = 320 - ps;
  int left = pj * step; if (left + ps > 320) left = 320 - ps;
  atomicAdd(&resid[((long)c * 320 + top + py) * 320 + left + px], xtok[i]);
}

__global__ void dmask_kernel(float* __restrict__ dm, int ps, int step)
{
  int p = blockIdx.x * blockDim.x + threadIdx.x;
  if (p >= 320) return;
  float v = 1.0f;
  for (int i = step; i < 320 + step - ps; i += step) {
    int top = i, down = i + ps - step;
    if (top + ps > 320) top = 320 - ps;
    if (p >= top && p < down) v *= 0.5f;
  }
  dm[p] = v;
}

__global__ void resid_scale_kernel(float* __restrict__ resid, const float* __restrict__ dm, long total)
{
  long i = (long)blockIdx.x * blockDim.x + threadIdx.x;
  if (i >= total) return;
  int x = (int)(i % 320);
  int y = (int)((i / 320) % 320);
  resid[i] *= dm[y] * dm[x];
}

// ---------------------------------------------------------------------------
// Convolutions / resize / pixel-shuffle tail
// ---------------------------------------------------------------------------
__global__ void conv3x3_kernel(const float* __restrict__ in, const float* __restrict__ w,
                               const float* __restrict__ bs, float* __restrict__ out,
                               int Cin, int add, long total)
{
  long i = (long)blockIdx.x * blockDim.x + threadIdx.x;
  if (i >= total) return;
  int x = (int)(i % 320);
  int y = (int)((i / 320) % 320);
  int co = (int)(i / (320 * 320));
  float s = bs[co];
  for (int ci = 0; ci < Cin; ++ci) {
    const float* wp = w + ((long)(co * Cin + ci)) * 9;
    const float* ip = in + (long)ci * 320 * 320;
    for (int dy = 0; dy < 3; ++dy) {
      int iy = y + dy - 1;
      if (iy < 0 || iy >= 320) continue;
      for (int dx = 0; dx < 3; ++dx) {
        int ix = x + dx - 1;
        if (ix < 0 || ix >= 320) continue;
        s += wp[dy * 3 + dx] * ip[iy * 320 + ix];
      }
    }
  }
  if (add) out[i] += s; else out[i] = s;
}

__global__ void bilinear_kernel(const float* __restrict__ x, float* __restrict__ out)
{
  long i = (long)blockIdx.x * blockDim.x + threadIdx.x;
  if (i >= 3L * 640 * 640) return;
  int X = (int)(i % 640);
  int Y = (int)((i / 640) % 640);
  int c = (int)(i / (640 * 640));
  float sy = (Y + 0.5f) * 0.5f - 0.5f;
  float sx = (X + 0.5f) * 0.5f - 0.5f;
  int y0 = (int)floorf(sy), x0 = (int)floorf(sx);
  float fy = sy - y0, fx = sx - x0;
  int y0c = y0 < 0 ? 0 : (y0 > 319 ? 319 : y0);
  int y1c = (y0 + 1) < 0 ? 0 : ((y0 + 1) > 319 ? 319 : y0 + 1);
  int x0c = x0 < 0 ? 0 : (x0 > 319 ? 319 : x0);
  int x1c = (x0 + 1) < 0 ? 0 : ((x0 + 1) > 319 ? 319 : x0 + 1);
  const float* p = x + (long)c * 320 * 320;
  float v00 = p[y0c * 320 + x0c], v01 = p[y0c * 320 + x1c];
  float v10 = p[y1c * 320 + x0c], v11 = p[y1c * 320 + x1c];
  out[i] = (1 - fy) * ((1 - fx) * v00 + fx * v01) + fy * ((1 - fx) * v10 + fx * v11);
}

// last conv: reads pixel_shuffle(upbuf,2) through leaky-relu, adds into d_out (base)
__global__ void lastconv_kernel(const float* __restrict__ up, const float* __restrict__ w,
                                const float* __restrict__ bs, float* __restrict__ out)
{
  long i = (long)blockIdx.x * blockDim.x + threadIdx.x;
  if (i >= 3L * 640 * 640) return;
  int x = (int)(i % 640);
  int y = (int)((i / 640) % 640);
  int co = (int)(i / (640 * 640));
  float s = bs[co];
  for (int ci = 0; ci < 40; ++ci) {
    const float* wp = w + ((long)(co * 40 + ci)) * 9;
    for (int dy = 0; dy < 3; ++dy) {
      int iy = y + dy - 1;
      if (iy < 0 || iy >= 640) continue;
      for (int dx = 0; dx < 3; ++dx) {
        int ix = x + dx - 1;
        if (ix < 0 || ix >= 640) continue;
        int ch = ci * 4 + (iy & 1) * 2 + (ix & 1);
        float v = up[((long)ch * 320 + (iy >> 1)) * 320 + (ix >> 1)];
        v = v >= 0.0f ? v : 0.1f * v;
        s += wp[dy * 3 + dx] * v;
      }
    }
  }
  out[i] += s;
}

// weight prep: fp32 [O][K] -> swizzled f16 panel [kt][ot][lane][16]
// (k = kt*32 + (lane>=16)*16 + e, o = ot*16 + (lane&15); zero-padded)
__global__ void prep_w_kernel(const float* __restrict__ src, half_t* __restrict__ dst,
                              int K, int O, int KT, int OT)
{
  int i = blockIdx.x * blockDim.x + threadIdx.x;
  if (i >= KT * OT * 512) return;
  int e    = i & 15;
  int lane = (i >> 4) & 31;
  int ot   = (i >> 9) % OT;
  int kt   = (i >> 9) / OT;
  int k = kt * 32 + ((lane >> 4) << 4) + e;
  int o = ot * 16 + (lane & 15);
  dst[i] = (k < K && o < O) ? (half_t)src[o * K + k] : (half_t)0.0f;
}

// ---------------------------------------------------------------------------
// Host orchestration
// ---------------------------------------------------------------------------
struct LayerP { const float *ln1_g,*ln1_b,*wq,*wk,*wv,*ln2_g,*ln2_b,*fc1_w,*fc1_b,*fc2_w,*fc2_b; };
struct BlockP { const float *match_g,*match_b,*mid_w,*mid_b; LayerP lay[3]; };
struct NetP   { const float *x,*first_w,*first_b,*up_w,*up_b,*last_w,*last_b; BlockP blk[8]; };

extern "C" void kernel_launch(void* const* d_in, const int* in_sizes, int n_in,
                              void* d_out, int out_size, void* d_ws, size_t ws_size,
                              hipStream_t stream)
{
  (void)out_size; (void)ws_size;
  const float* in[303];
  for (int i = 0; i < n_in && i < 303; ++i) in[i] = (const float*)d_in[i];

  NetP P;
  int i = 0;
  bool insertion = (n_in > 0 && in_sizes[0] == 3 * 320 * 320);
  if (insertion) {
    P.x = in[i++]; P.first_w = in[i++]; P.first_b = in[i++];
    P.up_w = in[i++]; P.up_b = in[i++]; P.last_w = in[i++]; P.last_b = in[i++];
    for (int b = 0; b < 8; ++b) {
      BlockP& B = P.blk[b];
      B.match_g = in[i++]; B.match_b = in[i++]; B.mid_w = in[i++]; B.mid_b = in[i++];
      for (int l = 0; l < 3; ++l) {
        LayerP& L = B.lay[l];
        L.ln1_g = in[i++]; L.ln1_b = in[i++]; L.wq = in[i++]; L.wk = in[i++]; L.wv = in[i++];
        L.ln2_g = in[i++]; L.ln2_b = in[i++]; L.fc1_w = in[i++]; L.fc1_b = in[i++];
        L.fc2_w = in[i++]; L.fc2_b = in[i++];
      }
    }
  } else {  // sorted-key pytree flatten
    for (int b = 0; b < 8; ++b) {
      BlockP& B = P.blk[b];
      for (int l = 0; l < 3; ++l) {
        LayerP& L = B.lay[l];
        L.fc1_b = in[i++]; L.fc1_w = in[i++]; L.fc2_b = in[i++]; L.fc2_w = in[i++];
        L.ln1_b = in[i++]; L.ln1_g = in[i++]; L.ln2_b = in[i++]; L.ln2_g = in[i++];
        L.wk = in[i++]; L.wq = in[i++]; L.wv = in[i++];
      }
      B.match_b = in[i++]; B.match_g = in[i++]; B.mid_b = in[i++]; B.mid_w = in[i++];
    }
    P.first_b = in[i++]; P.first_w = in[i++]; P.last_b = in[i++]; P.last_w = in[i++];
    P.up_b = in[i++]; P.up_w = in[i++]; P.x = in[i++];
  }

  // ---- workspace layout (8KB guard pads) ----
  char* ws = (char*)d_ws;
  size_t off = 0;
  auto alloc = [&](size_t bytes) { size_t r = off; off += (bytes + 8191) & ~(size_t)255; return r; };
  const long FE = 40L * 320 * 320;
  const long TT = 147456;  // max tokens per block (ps=12: 1024 patches x 144)
  size_t oF    = alloc(FE * 4);
  size_t oRes  = alloc(FE * 4);
  size_t oX    = alloc(TT * 40 * 4);
  size_t oYH   = alloc(TT * 64 * 2);
  size_t oXNH  = alloc(TT * 64 * 2);
  size_t oQH   = alloc(TT * 32 * 2);
  size_t oKH   = alloc(TT * 32 * 2);
  size_t oVH   = alloc(TT * 48 * 2);
  size_t oVS   = alloc(16L * 1024 * 1024);  // swizzled V (max 1024*5*1536*2 = 15.7MB)
  size_t oH1   = alloc(TT * 96 * 2);
  size_t oMean = alloc(1024 * 40 * 4);
  size_t oQln  = alloc(1024 * 40 * 4);
  size_t oIdx  = alloc(1024 * 4);
  size_t oDm   = alloc(320 * 4);
  size_t oW    = alloc(24L * 17920 * 2);
  size_t oUp   = oX;  // up-conv buffer (65.5MB) reuses token region after transformer stage

  float*  fbuf  = (float*)(ws + oF);
  float*  resid = (float*)(ws + oRes);
  float*  xtok  = (float*)(ws + oX);
  half_t* yh    = (half_t*)(ws + oYH);
  half_t* xnh   = (half_t*)(ws + oXNH);
  half_t* qh    = (half_t*)(ws + oQH);
  half_t* kh    = (half_t*)(ws + oKH);
  half_t* vh    = (half_t*)(ws + oVH);
  half_t* vswz  = (half_t*)(ws + oVS);
  half_t* h1h   = (half_t*)(ws + oH1);
  float*  means = (float*)(ws + oMean);
  float*  qln   = (float*)(ws + oQln);
  int*    idxb  = (int*)(ws + oIdx);
  float*  dmask = (float*)(ws + oDm);
  half_t* wbase = (half_t*)(ws + oW);
  float*  upbuf = (float*)(ws + oUp);
  float*  outp  = (float*)d_out;

  auto g1 = [](long total) { return dim3((unsigned)((total + 255) / 256)); };

  // ---- swizzled f16 weight panels ----
  for (int b = 0; b < 8; ++b)
    for (int l = 0; l < 3; ++l) {
      const LayerP& L = P.blk[b].lay[l];
      half_t* wl = wbase + (size_t)(b * 3 + l) * 17920;
      prep_w_kernel<<<g1(2 * 2 * 512), 256, 0, stream>>>(L.wq,    wl + 0,     40, 24, 2, 2);
      prep_w_kernel<<<g1(2 * 2 * 512), 256, 0, stream>>>(L.wk,    wl + 2048,  40, 24, 2, 2);
      prep_w_kernel<<<g1(2 * 3 * 512), 256, 0, stream>>>(L.wv,    wl + 4096,  40, 40, 2, 3);
      prep_w_kernel<<<g1(2 * 6 * 512), 256, 0, stream>>>(L.fc1_w, wl + 7168,  40, 72, 2, 6);
      prep_w_kernel<<<g1(3 * 3 * 512), 256, 0, stream>>>(L.fc2_w, wl + 13312, 72, 40, 3, 3);
    }

  // ---- base (bilinear x2) straight into d_out ----
  bilinear_kernel<<<g1(3L * 640 * 640), 256, 0, stream>>>(P.x, outp);

  // ---- first conv: f = conv3x3(x) ----
  conv3x3_kernel<<<g1(FE), 256, 0, stream>>>(P.x, P.first_w, P.first_b, fbuf, 3, 0, FE);

  static const int PSL[8] = {12, 16, 20, 24, 12, 16, 20, 24};
  for (int b = 0; b < 8; ++b) {
    int ps = PSL[b], step = ps - 2;
    int nh = (318 + step - 1) / step;
    int n = nh * nh, T = ps * ps;
    int chunks = (T + 31) / 32;
    long nT = (long)n * T;
    long tt = nT / 16;

    patch_divide_kernel<<<g1(nT * 40), 256, 0, stream>>>(fbuf, xtok, ps, step, nh, nT * 40);
    patch_mean_kernel<<<g1(n * 40), 256, 0, stream>>>(xtok, means, T, n * 40);
    match_ln_kernel<<<g1(n), 256, 0, stream>>>(means, P.blk[b].match_g, P.blk[b].match_b, qln, n);
    match_argmax_kernel<<<dim3((unsigned)n), 128, 0, stream>>>(qln, idxb, n);
    gather_y_kernel<<<g1(nT * 64), 256, 0, stream>>>(xtok, idxb, yh, T, nT * 64);

    for (int l = 0; l < 3; ++l) {
      const LayerP& L = P.blk[b].lay[l];
      half_t* wl = wbase + (size_t)(b * 3 + l) * 17920;
      ln_kernel<<<g1(nT), 256, 0, stream>>>(xtok, L.ln1_g, L.ln1_b, xnh, nT);
      gemm_tok_kernel<<<dim3((unsigned)(tt * 2)), 32, 0, stream>>>(
          xnh, 64, wl + 0, nullptr, qh, 32, nullptr, 0, nT, 2, 24, 2, 0);
      const half_t* kv = (l == 0) ? yh : xnh;
      gemm_tok_kernel<<<dim3((unsigned)(tt * 2)), 32, 0, stream>>>(
          kv, 64, wl + 2048, nullptr, kh, 32, nullptr, 0, nT, 2, 24, 2, 0);
      gemm_tok_kernel<<<dim3((unsigned)(tt * 3)), 32, 0, stream>>>(
          kv, 64, wl + 4096, nullptr, vh, 48, nullptr, 0, nT, 2, 40, 3, 0);
      long vstot = (long)n * chunks * 1536;
      repack_v_kernel<<<g1(vstot), 256, 0, stream>>>(vh, vswz, T, chunks, vstot);
      attn_kernel<<<dim3((unsigned)tt), 32, 0, stream>>>(qh, kh, vswz, xtok, T, chunks);
      ln_kernel<<<g1(nT), 256, 0, stream>>>(xtok, L.ln2_g, L.ln2_b, xnh, nT);
      gemm_tok_kernel<<<dim3((unsigned)(tt * 6)), 32, 0, stream>>>(
          xnh, 64, wl + 7168, L.fc1_b, h1h, 96, nullptr, 0, nT, 2, 72, 6, 1);
      gemm_tok_kernel<<<dim3((unsigned)(tt * 3)), 32, 0, stream>>>(
          h1h, 96, wl + 13312, L.fc2_b, nullptr, 0, xtok, 40, nT, 3, 40, 3, 0);
    }

    hipMemsetAsync(resid, 0, FE * 4, stream);
    patch_scatter_kernel<<<g1(nT * 40), 256, 0, stream>>>(xtok, resid, ps, step, nh, nT * 40);
    dmask_kernel<<<2, 256, 0, stream>>>(dmask, ps, step);
    resid_scale_kernel<<<g1(FE), 256, 0, stream>>>(resid, dmask, FE);
    conv3x3_kernel<<<g1(FE), 256, 0, stream>>>(resid, P.blk[b].mid_w, P.blk[b].mid_b, fbuf, 40, 1, FE);
  }

  // ---- tail: up conv -> pixel shuffle + leaky relu + last conv, added onto base ----
  conv3x3_kernel<<<g1(160L * 320 * 320), 256, 0, stream>>>(fbuf, P.up_w, P.up_b, upbuf, 40, 0, 160L * 320 * 320);
  lastconv_kernel<<<g1(3L * 640 * 640), 256, 0, stream>>>(upbuf, P.last_w, P.last_b, outp);
}
